// MambaBlock_30872224923989
// MI455X (gfx1250) — matrix operand order repaired and checked by reading, not compiled, not run on hardware
//
#include <hip/hip_runtime.h>

// ---------------- problem constants (from reference) ----------------
#define NEMBD   1024
#define DSTATE  16
#define DCONV   4
#define DINNER  2048
#define DTRANK  64
#define BATCH   2
#define SEQLEN  2048
#define BL      (BATCH * SEQLEN)      // 4096 flattened rows
#define XDBLW   (DTRANK + 2 * DSTATE) // 96
#define XDBL_NPAD 128                 // W_x rows padded so GEMM loads need no guard

// segmented scan
#define NSEG    16
#define LSEG    (SEQLEN / NSEG)       // 128

// GEMM tiling
#define BM 128                        // block M tile (8 waves x 16)
#define BN 64                         // block N tile
#define BK 32                         // K step (one bf16 WMMA); nK = K/BK must be EVEN
#define BKP 40                        // LDS row pitch in bf16 (80 B: 16B-aligned, de-conflicted)

// ---------------- types for CDNA5 WMMA ----------------
typedef __bf16 bf16_t;
typedef __attribute__((ext_vector_type(16))) __bf16 v16bf;
typedef __attribute__((ext_vector_type(8)))  __bf16 v8bf;
typedef __attribute__((ext_vector_type(8)))  float  v8f;
typedef __attribute__((ext_vector_type(4)))  int    v4i;
// address-space qualified int4 for the async global->LDS builtin
typedef __attribute__((address_space(1))) v4i v4i_g;
typedef __attribute__((address_space(3))) v4i v4i_l;

union Frag {
    v16bf v;
    v8bf  h[2];
};

// gfx1250 async global->LDS path (guarded: falls back to load+ds_store)
#if defined(__has_builtin)
#if __has_builtin(__builtin_amdgcn_global_load_async_to_lds_b128) && \
    __has_builtin(__builtin_amdgcn_s_wait_asynccnt)
#define USE_ASYNC_LDS 1
#endif
#endif

__device__ __forceinline__ float silu_f(float x) {
    return x / (1.0f + __expf(-x));
}
__device__ __forceinline__ float softplus_f(float x) {
    return (x > 20.0f) ? x : log1pf(__expf(x));
}

// ---------------- fp32 -> bf16 pack ----------------
__global__ void cvt_bf16_kernel(const float* __restrict__ src,
                                bf16_t* __restrict__ dst, int n) {
    int i = blockIdx.x * blockDim.x + threadIdx.x;
    if (i < n) dst[i] = (bf16_t)src[i];
}

// fp32 -> bf16 with zero row padding (rows >= srcRows become 0)
__global__ void cvt_bf16_pad_rows_kernel(const float* __restrict__ src,
                                         bf16_t* __restrict__ dst,
                                         int srcRows, int dstRows, int K) {
    int i = blockIdx.x * blockDim.x + threadIdx.x;
    if (i >= dstRows * K) return;
    const int r = i / K;
    dst[i] = (r < srcRows) ? (bf16_t)src[i] : (bf16_t)0.0f;
}

// ---------------- WMMA GEMM: C[M,ldc] = A[M,K] @ Bw[Npad,K]^T ----------------
// Launcher guarantees: M % BM == 0, (K/BK) even, Bw padded to ceil(N/BN)*BN
// rows -> NO guards in the K-loop.  epilogue: 0 = plain, 1 = softplus(x+bias)
__global__ __launch_bounds__(256)
void gemm_wmma_bf16(const bf16_t* __restrict__ A,
                    const bf16_t* __restrict__ Bw,
                    float* __restrict__ C,
                    const float* __restrict__ bias,
                    int M, int N, int K, int ldc, int epilogue, int grid_n) {
    __shared__ bf16_t sB[2][BN * BKP];              // double-buffered B tile (10 KB)

    const int tid   = threadIdx.x;
    const int wave  = tid >> 5;
    const int lane  = tid & 31;
    const int bm    = (int)blockIdx.x / grid_n;
    const int bn    = (int)blockIdx.x % grid_n;
    const int m0    = bm * BM + wave * 16;
    const int n0    = bn * BN;
    const int lhalf = lane >> 4;                    // 0: lanes 0-15, 1: lanes 16-31
    const int lrow  = lane & 15;
    const int nK    = K / BK;                       // even by construction

    // B staging map: each of 256 threads moves one 16-byte chunk per K-step
    const int brow = tid >> 2;                      // 0..63
    const int bkc  = (tid & 3) * 8;                 // 0,8,16,24
    const bf16_t* bsrc = Bw + (size_t)(n0 + brow) * K + bkc;
    bf16_t* bdst = &sB[0][brow * BKP + bkc];

    // A fragment base (ISA 16-bit A 16x32 layout: half-lane picks K chunks {0,16} / {8,24})
    const bf16_t* abase = A + (size_t)(m0 + lrow) * K + lhalf * 8;

    v8f acc[4] = {};

#ifdef USE_ASYNC_LDS
#define STAGE_B(kbi, buf)                                                         \
    __builtin_amdgcn_global_load_async_to_lds_b128(                               \
        (v4i_g*)(bsrc + (size_t)(kbi) * BK),                                      \
        (v4i_l*)(bdst + (buf) * (BN * BKP)),                                      \
        0, 0)
#define WAIT_STAGE() __builtin_amdgcn_s_wait_asynccnt(0)
#else
#define STAGE_B(kbi, buf)                                                         \
    *(v8bf*)(bdst + (buf) * (BN * BKP)) = *(const v8bf*)(bsrc + (size_t)(kbi) * BK)
#define WAIT_STAGE()
#endif

#define LOAD_A(FA, kbi)                                                           \
    {                                                                             \
        const bf16_t* ap = abase + (size_t)(kbi) * BK;                            \
        (FA).h[0] = *(const v8bf*)(ap);                                           \
        (FA).h[1] = *(const v8bf*)(ap + 16);                                      \
    }

// Issue all 8 ds_load_b128 first (in-order returns -> progressive dscnt waits),
// then 4 back-to-back WMMAs on distinct accumulators.
#define COMPUTE(FA, BUF)                                                          \
    {                                                                             \
        const bf16_t* bb = &sB[BUF][lrow * BKP + lhalf * 8];                      \
        Frag fb0, fb1, fb2, fb3;                                                  \
        fb0.h[0] = *(const v8bf*)(bb);                                            \
        fb0.h[1] = *(const v8bf*)(bb + 16);                                       \
        fb1.h[0] = *(const v8bf*)(bb + 16 * BKP);                                 \
        fb1.h[1] = *(const v8bf*)(bb + 16 * BKP + 16);                            \
        fb2.h[0] = *(const v8bf*)(bb + 32 * BKP);                                 \
        fb2.h[1] = *(const v8bf*)(bb + 32 * BKP + 16);                            \
        fb3.h[0] = *(const v8bf*)(bb + 48 * BKP);                                 \
        fb3.h[1] = *(const v8bf*)(bb + 48 * BKP + 16);                            \
        acc[0] = __builtin_amdgcn_wmma_f32_16x16x32_bf16(                         \
            false, (FA).v, false, fb0.v, (short)0, acc[0], false, false);         \
        acc[1] = __builtin_amdgcn_wmma_f32_16x16x32_bf16(                         \
            false, (FA).v, false, fb1.v, (short)0, acc[1], false, false);         \
        acc[2] = __builtin_amdgcn_wmma_f32_16x16x32_bf16(                         \
            false, (FA).v, false, fb2.v, (short)0, acc[2], false, false);         \
        acc[3] = __builtin_amdgcn_wmma_f32_16x16x32_bf16(                         \
            false, (FA).v, false, fb3.v, (short)0, acc[3], false, false);         \
    }

    // prologue
    STAGE_B(0, 0);
    Frag fa0, fa1;
    LOAD_A(fa0, 0);
    WAIT_STAGE();
    __syncthreads();

    // ping-pong over K-step pairs (nK even): no fragment copies, no parity math
    for (int kbi = 0; kbi < nK; kbi += 2) {
        // even step: compute from buf0/fa0 while staging buf1/fa1
        STAGE_B(kbi + 1, 1);
        LOAD_A(fa1, kbi + 1);
        COMPUTE(fa0, 0);
        WAIT_STAGE();
        __syncthreads();

        // odd step: compute from buf1/fa1 while staging buf0/fa0 for next pair
        const bool more = (kbi + 2 < nK);
        if (more) {
            STAGE_B(kbi + 2, 0);
            LOAD_A(fa0, kbi + 2);
        }
        COMPUTE(fa1, 1);
        if (more) {
            WAIT_STAGE();
            __syncthreads();
        }
    }

#pragma unroll
    for (int nt = 0; nt < 4; ++nt) {
        const int n = n0 + nt * 16 + lrow;
        if (n >= N) continue;                       // only x_dbl (N=96) hits this
        const float bv = bias ? bias[n] : 0.0f;
#pragma unroll
        for (int e = 0; e < 8; ++e) {
            const int m = m0 + e + lhalf * 8;
            float v = acc[nt][e];
            if (epilogue == 1) v = softplus_f(v + bv);
            C[(size_t)m * ldc + n] = v;
        }
    }
#undef STAGE_B
#undef WAIT_STAGE
#undef LOAD_A
#undef COMPUTE
}

// ---------------- causal depthwise conv1d + bias + silu ----------------
__global__ void conv_silu_kernel(const float* __restrict__ xr,
                                 const float* __restrict__ cw,
                                 const float* __restrict__ cb,
                                 float* __restrict__ xc,
                                 bf16_t* __restrict__ xcb) {
    int idx = blockIdx.x * blockDim.x + threadIdx.x;
    if (idx >= BL * DINNER) return;
    const int d = idx & (DINNER - 1);
    const int t = (idx / DINNER) & (SEQLEN - 1);
    const int b = idx / (DINNER * SEQLEN);
    const float* w = cw + d * DCONV;
    float acc = cb[d];
#pragma unroll
    for (int k = 0; k < DCONV; ++k) {
        const int tt = t - (DCONV - 1) + k;
        if (tt >= 0)
            acc += w[k] * xr[((size_t)(b * SEQLEN + tt)) * (2 * DINNER) + d];
    }
    const float s = silu_f(acc);
    xc[idx]  = s;
    xcb[idx] = (bf16_t)s;
}

// ---------------- slice dt-low (first 64 cols of x_dbl) to bf16 ----------------
__global__ void pack_dtlow_kernel(const float* __restrict__ xdbl,
                                  bf16_t* __restrict__ dtl) {
    int i = blockIdx.x * blockDim.x + threadIdx.x;
    if (i >= BL * DTRANK) return;
    const int r = i >> 6, j = i & 63;
    dtl[i] = (bf16_t)xdbl[(size_t)r * XDBLW + j];
}

// =============== segmented (3-pass) selective scan ===============
// h_t = a_t*h_{t-1} + u_t is associative -> NSEG segments of LSEG steps.
// Pass1: per-segment decay product P and zero-init final F (16x parallelism).
// Pass2: tiny sequential combine over segments. Pass3: replay with corrected
// initial state, emit y.

__global__ __launch_bounds__(256)
void scan_pass1(const float* __restrict__ xc, const float* __restrict__ dtsp,
                const float* __restrict__ xdbl, const float* __restrict__ A_log,
                float* __restrict__ Pseg, float* __restrict__ Fseg) {
    const int b   = (int)blockIdx.x / (NSEG * 8);
    const int seg = ((int)blockIdx.x / 8) % NSEG;
    const int d   = ((int)blockIdx.x & 7) * 256 + threadIdx.x;

    __shared__ float sB[LSEG][DSTATE];              // 8 KB: B for whole segment
    const float* bbase = xdbl + ((size_t)(b * SEQLEN + seg * LSEG)) * XDBLW + DTRANK;
    for (int i = threadIdx.x; i < LSEG * DSTATE; i += 256) {
        const int t = i >> 4, n = i & (DSTATE - 1);
        sB[t][n] = bbase[(size_t)t * XDBLW + n];
    }
    __syncthreads();

    float a[DSTATE], P[DSTATE], F[DSTATE];
#pragma unroll
    for (int n = 0; n < DSTATE; ++n) {
        a[n] = -__expf(A_log[(size_t)d * DSTATE + n]);
        P[n] = 1.0f;
        F[n] = 0.0f;
    }
    for (int tt = 0; tt < LSEG; ++tt) {
        const size_t off = ((size_t)(b * SEQLEN + seg * LSEG + tt)) * DINNER + d;
        const float dt = dtsp[off];
        const float dx = dt * xc[off];
#pragma unroll
        for (int n = 0; n < DSTATE; ++n) {
            const float e = __expf(dt * a[n]);
            P[n] *= e;
            F[n] = e * F[n] + dx * sB[tt][n];
        }
    }
    const size_t o = (((size_t)seg * BATCH + b) * DINNER + d) * DSTATE;
#pragma unroll
    for (int n = 0; n < DSTATE; ++n) {
        Pseg[o + n] = P[n];
        Fseg[o + n] = F[n];
    }
}

__global__ void scan_pass2(const float* __restrict__ Pseg,
                           const float* __restrict__ Fseg,
                           const float* __restrict__ h0,
                           float* __restrict__ Hinit, float* __restrict__ hout) {
    int i = blockIdx.x * blockDim.x + threadIdx.x;
    if (i >= BATCH * DINNER * DSTATE) return;
    const int b   = i / (DINNER * DSTATE);
    const int rem = i % (DINNER * DSTATE);
    float h = h0[i];                                // h0 flat layout matches (B,D,N)
    for (int s = 0; s < NSEG; ++s) {
        const size_t o = ((size_t)s * BATCH + b) * (DINNER * DSTATE) + rem;
        Hinit[o] = h;
        h = Fseg[o] + Pseg[o] * h;
    }
    hout[i] = h;                                    // final carry -> d_out tail
}

__global__ __launch_bounds__(256)
void scan_pass3(const float* __restrict__ xc, const float* __restrict__ dtsp,
                const float* __restrict__ xdbl, const float* __restrict__ A_log,
                const float* __restrict__ Hinit, float* __restrict__ ys) {
    const int b   = (int)blockIdx.x / (NSEG * 8);
    const int seg = ((int)blockIdx.x / 8) % NSEG;
    const int d   = ((int)blockIdx.x & 7) * 256 + threadIdx.x;

    __shared__ float sBC[LSEG][2 * DSTATE];         // 16 KB: B and C for segment
    const float* bbase = xdbl + ((size_t)(b * SEQLEN + seg * LSEG)) * XDBLW + DTRANK;
    for (int i = threadIdx.x; i < LSEG * 2 * DSTATE; i += 256) {
        const int t = i >> 5, j = i & 31;
        sBC[t][j] = bbase[(size_t)t * XDBLW + j];
    }
    __syncthreads();

    float a[DSTATE], h[DSTATE];
    const size_t o = (((size_t)seg * BATCH + b) * DINNER + d) * DSTATE;
#pragma unroll
    for (int n = 0; n < DSTATE; ++n) {
        a[n] = -__expf(A_log[(size_t)d * DSTATE + n]);
        h[n] = Hinit[o + n];
    }
    for (int tt = 0; tt < LSEG; ++tt) {
        const size_t off = ((size_t)(b * SEQLEN + seg * LSEG + tt)) * DINNER + d;
        const float dt = dtsp[off];
        const float dx = dt * xc[off];
        float y = 0.0f;
#pragma unroll
        for (int n = 0; n < DSTATE; ++n) {
            h[n] = __expf(dt * a[n]) * h[n] + dx * sBC[tt][n];
            y += h[n] * sBC[tt][DSTATE + n];
        }
        ys[off] = y;
    }
}

// ---------------- gate: yg = bf16((ys + xc*D) * silu(res)) ----------------
__global__ void gate_kernel(const float* __restrict__ ys,
                            const float* __restrict__ xc,
                            const float* __restrict__ Dp,
                            const float* __restrict__ xr,
                            bf16_t* __restrict__ yg) {
    int i = blockIdx.x * blockDim.x + threadIdx.x;
    if (i >= BL * DINNER) return;
    const int d   = i & (DINNER - 1);
    const int row = i / DINNER;
    const float res = xr[(size_t)row * (2 * DINNER) + DINNER + d];
    yg[i] = (bf16_t)((ys[i] + xc[i] * Dp[d]) * silu_f(res));
}

// =====================================================================
extern "C" void kernel_launch(void* const* d_in, const int* in_sizes, int n_in,
                              void* d_out, int out_size, void* d_ws, size_t ws_size,
                              hipStream_t stream) {
    (void)in_sizes; (void)n_in; (void)out_size; (void)ws_size;

    const float* x      = (const float*)d_in[0];
    const float* h0     = (const float*)d_in[1];
    const float* W_in   = (const float*)d_in[2];
    const float* conv_w = (const float*)d_in[3];
    const float* conv_b = (const float*)d_in[4];
    const float* W_x    = (const float*)d_in[5];
    const float* W_dt   = (const float*)d_in[6];
    const float* b_dt   = (const float*)d_in[7];
    const float* A_log  = (const float*)d_in[8];
    const float* Dp     = (const float*)d_in[9];
    const float* W_out  = (const float*)d_in[10];

    float* out  = (float*)d_out;                        // (BL, NEMBD)
    float* hout = out + (size_t)BL * NEMBD;             // (BATCH, DINNER, DSTATE)

    // ---- workspace bump allocator (256B aligned) ----
    char*  ws  = (char*)d_ws;
    size_t cur = 0;
    auto alloc = [&](size_t bytes) -> void* {
        void* p = ws + cur;
        cur += (bytes + 255) & ~(size_t)255;
        return p;
    };
    bf16_t* xb    = (bf16_t*)alloc((size_t)BL * NEMBD * 2);
    bf16_t* Winb  = (bf16_t*)alloc((size_t)2 * DINNER * NEMBD * 2);
    bf16_t* Wxb   = (bf16_t*)alloc((size_t)XDBL_NPAD * DINNER * 2);   // padded rows
    bf16_t* Wdtb  = (bf16_t*)alloc((size_t)DINNER * DTRANK * 2);
    bf16_t* Woutb = (bf16_t*)alloc((size_t)NEMBD * DINNER * 2);
    float*  xr    = (float*) alloc((size_t)BL * 2 * DINNER * 4);
    float*  xc    = (float*) alloc((size_t)BL * DINNER * 4);
    bf16_t* xcb   = (bf16_t*)alloc((size_t)BL * DINNER * 2);
    float*  xdbl  = (float*) alloc((size_t)BL * XDBLW * 4);
    bf16_t* dtl   = (bf16_t*)alloc((size_t)BL * DTRANK * 2);
    float*  dtsp  = (float*) alloc((size_t)BL * DINNER * 4);
    float*  ysb   = (float*) alloc((size_t)BL * DINNER * 4);
    bf16_t* yg    = (bf16_t*)alloc((size_t)BL * DINNER * 2);
    const size_t segN = (size_t)NSEG * BATCH * DINNER * DSTATE;       // 1M floats
    float*  Pseg  = (float*)alloc(segN * 4);
    float*  Fseg  = (float*)alloc(segN * 4);
    float*  Hinit = (float*)alloc(segN * 4);

    const int TB = 256;
    auto nblk = [](size_t n, int tb) { return (unsigned)((n + tb - 1) / tb); };

    // ---- pack fp32 operands to bf16 ----
    cvt_bf16_kernel<<<nblk((size_t)BL * NEMBD, TB), TB, 0, stream>>>(x, xb, BL * NEMBD);
    cvt_bf16_kernel<<<nblk((size_t)2 * DINNER * NEMBD, TB), TB, 0, stream>>>(W_in, Winb, 2 * DINNER * NEMBD);
    cvt_bf16_pad_rows_kernel<<<nblk((size_t)XDBL_NPAD * DINNER, TB), TB, 0, stream>>>(W_x, Wxb, XDBLW, XDBL_NPAD, DINNER);
    cvt_bf16_kernel<<<nblk((size_t)DINNER * DTRANK, TB), TB, 0, stream>>>(W_dt, Wdtb, DINNER * DTRANK);
    cvt_bf16_kernel<<<nblk((size_t)NEMBD * DINNER, TB), TB, 0, stream>>>(W_out, Woutb, NEMBD * DINNER);

    // ---- 1) in_proj: xr = x @ W_in^T  (M=4096, N=4096, K=1024) ----
    {
        const int gn = (2 * DINNER) / BN;
        gemm_wmma_bf16<<<(BL / BM) * gn, TB, 0, stream>>>(
            xb, Winb, xr, nullptr, BL, 2 * DINNER, NEMBD, 2 * DINNER, 0, gn);
    }
    // ---- 2) depthwise causal conv + silu ----
    conv_silu_kernel<<<nblk((size_t)BL * DINNER, TB), TB, 0, stream>>>(xr, conv_w, conv_b, xc, xcb);

    // ---- 3) x_dbl = xc @ W_x^T  (M=4096, N=96 (rows padded to 128), K=2048) ----
    {
        const int gn = XDBL_NPAD / BN;
        gemm_wmma_bf16<<<(BL / BM) * gn, TB, 0, stream>>>(
            xcb, Wxb, xdbl, nullptr, BL, XDBLW, DINNER, XDBLW, 0, gn);
    }
    // ---- 4) dt = softplus(dt_low @ W_dt^T + b_dt)  (M=4096, N=2048, K=64) ----
    pack_dtlow_kernel<<<nblk((size_t)BL * DTRANK, TB), TB, 0, stream>>>(xdbl, dtl);
    {
        const int gn = DINNER / BN;
        gemm_wmma_bf16<<<(BL / BM) * gn, TB, 0, stream>>>(
            dtl, Wdtb, dtsp, b_dt, BL, DINNER, DTRANK, DINNER, 1, gn);
    }
    // ---- 5) segmented selective scan ----
    scan_pass1<<<BATCH * NSEG * (DINNER / 256), TB, 0, stream>>>(xc, dtsp, xdbl, A_log, Pseg, Fseg);
    scan_pass2<<<nblk((size_t)BATCH * DINNER * DSTATE, TB), TB, 0, stream>>>(Pseg, Fseg, h0, Hinit, hout);
    scan_pass3<<<BATCH * NSEG * (DINNER / 256), TB, 0, stream>>>(xc, dtsp, xdbl, A_log, Hinit, ysb);

    // ---- 6) gate ----
    gate_kernel<<<nblk((size_t)BL * DINNER, TB), TB, 0, stream>>>(ysb, xc, Dp, xr, yg);

    // ---- 7) out = yg @ W_out^T  (M=4096, N=1024, K=2048) -> d_out ----
    {
        const int gn = NEMBD / BN;
        gemm_wmma_bf16<<<(BL / BM) * gn, TB, 0, stream>>>(
            yg, Woutb, out, nullptr, BL, NEMBD, DINNER, NEMBD, 0, gn);
    }
}
